// DecoderBlock_22454089024262
// MI455X (gfx1250) — compile-verified, matrix operand order checked by reading, not run on hardware
//
#include <hip/hip_runtime.h>

// ---------------------------------------------------------------------------
// Transformer decoder block (pre-LN, causal attention, ReLU FFN) for gfx1250.
// bf16 WMMA (v_wmma_f32_16x16x32_bf16) for all GEMM-shaped work, fp32 acc.
// GEMMs: 128x128 block tile, 64-deep K slabs, double-buffered LDS staging via
// GLOBAL_LOAD_ASYNC_TO_LDS_B128 (ASYNCcnt), 64x32 wave tiles (8 WMMA/k-step).
// ---------------------------------------------------------------------------

constexpr int Bc   = 2;
constexpr int Sc   = 2048;
constexpr int Dc   = 1024;
constexpr int Hc   = 16;
constexpr int DHc  = 64;       // Dc / Hc
constexpr int DFFc = 4096;
constexpr int Mrows = Bc * Sc; // 4096 token rows

typedef __attribute__((ext_vector_type(16))) __bf16 v16bf;
typedef __attribute__((ext_vector_type(8)))  __bf16 v8bf;
typedef __attribute__((ext_vector_type(8)))  float  v8f;

// ---- helpers --------------------------------------------------------------

__device__ inline unsigned short bf16bits(float f) {
    union { float f; unsigned int u; } a; a.f = f;
    unsigned int r = (a.u + 0x7FFFu + ((a.u >> 16) & 1u)) >> 16;  // RNE
    return (unsigned short)r;
}

__device__ inline __bf16 f2bf(float f) {
    union { unsigned short u; __bf16 b; } o; o.u = bf16bits(f);
    return o.b;
}

__device__ inline v8f wmma_bf16(v16bf a, v16bf b, v8f c) {
    // D = A(16x32) * B(32x16) + C(16x16), fp32 accumulate
    return __builtin_amdgcn_wmma_f32_16x16x32_bf16(
        false, a, false, b, (short)0, c, false, false);
}

// gfx1250 async global->LDS copy, 16 bytes per lane. ASYNCcnt-tracked.
// the LDS address is taken from the vdst VGPR; flat shared pointers keep the LDS offset in
// the low 32 bits, so we pass that directly.
__device__ inline void async_copy_b128(unsigned lds_off, const void* gptr) {
    asm volatile("global_load_async_to_lds_b128 %0, %1, off"
                 :: "v"(lds_off),
                    "v"((unsigned long long)(__SIZE_TYPE__)gptr)
                 : "memory");
}

template <int N>
__device__ inline void wait_asynccnt() {
    asm volatile("s_wait_asynccnt %0" :: "i"(N) : "memory");
}

__device__ inline unsigned lds_offset_of(const void* shared_ptr) {
    return (unsigned)(unsigned long long)(__SIZE_TYPE__)shared_ptr;
}

// Load one 16x32 (rows x k) WMMA operand tile in native lane layout for an
// NT GEMM: lane L(0..15) holds row (row0+L), k-chunks [k0..k0+7] and
// [k0+16..k0+23]; lane L(16..31) holds row (row0+L-16), chunks [k0+8..k0+15]
// and [k0+24..k0+31]. Two 16-byte loads per lane.
__device__ inline v16bf load_nt(const __bf16* __restrict__ p, int row0,
                                int ld, int k0, int lane) {
    int r  = row0 + (lane & 15);
    int kb = k0 + ((lane >> 4) << 3);
    const __bf16* q = p + (size_t)r * ld + kb;
    v8bf lo = *(const v8bf*)(q);
    v8bf hi = *(const v8bf*)(q + 16);
    return __builtin_shufflevector(lo, hi, 0,1,2,3,4,5,6,7,8,9,10,11,12,13,14,15);
}

// Operand fetch from an LDS slab with row stride 64 halves (K-slab depth 64).
// ks selects the 32-wide k-step within the slab.
__device__ inline v16bf lds_tile64(const __bf16* base, int row0, int ks,
                                   int lane) {
    int r  = row0 + (lane & 15);
    int kb = ks * 32 + ((lane >> 4) << 3);
    const __bf16* q = base + r * 64 + kb;
    v8bf lo = *(const v8bf*)(q);
    v8bf hi = *(const v8bf*)(q + 16);
    return __builtin_shufflevector(lo, hi, 0,1,2,3,4,5,6,7,8,9,10,11,12,13,14,15);
}

__device__ inline float rowmax16(float v) {
#pragma unroll
    for (int m = 1; m < 16; m <<= 1) v = fmaxf(v, __shfl_xor(v, m, 32));
    return v;
}
__device__ inline float rowsum16(float v) {
#pragma unroll
    for (int m = 1; m < 16; m <<= 1) v += __shfl_xor(v, m, 32);
    return v;
}

// ---- fp32 -> bf16 weight/activation convert -------------------------------

__global__ void cvt_kernel(const float* __restrict__ in, __bf16* __restrict__ out,
                           int n) {
    for (int i = blockIdx.x * blockDim.x + threadIdx.x; i < n;
         i += gridDim.x * blockDim.x)
        out[i] = f2bf(in[i]);
}

// ---- LayerNorm over D=1024, fp32 in -> bf16 out ---------------------------

__global__ __launch_bounds__(256)
void ln_kernel(const float* __restrict__ in, const float* __restrict__ g,
               const float* __restrict__ be, __bf16* __restrict__ out) {
    __shared__ float ssum[256];
    __shared__ float ssq[256];
    const int row = blockIdx.x;
    const int tid = threadIdx.x;
    const float4 v = ((const float4*)(in + (size_t)row * Dc))[tid];
    ssum[tid] = v.x + v.y + v.z + v.w;
    ssq[tid]  = v.x * v.x + v.y * v.y + v.z * v.z + v.w * v.w;
    __syncthreads();
#pragma unroll
    for (int o = 128; o > 0; o >>= 1) {
        if (tid < o) { ssum[tid] += ssum[tid + o]; ssq[tid] += ssq[tid + o]; }
        __syncthreads();
    }
    const float mu   = ssum[0] * (1.0f / Dc);
    const float var  = ssq[0] * (1.0f / Dc) - mu * mu;
    const float rstd = rsqrtf(var + 1e-5f);
    const float4 gv = ((const float4*)g)[tid];
    const float4 bv = ((const float4*)be)[tid];
    __bf16* o = out + (size_t)row * Dc + tid * 4;
    o[0] = f2bf((v.x - mu) * rstd * gv.x + bv.x);
    o[1] = f2bf((v.y - mu) * rstd * gv.y + bv.y);
    o[2] = f2bf((v.z - mu) * rstd * gv.z + bv.z);
    o[3] = f2bf((v.w - mu) * rstd * gv.w + bv.w);
}

// ---- NT GEMM: C[M,N] = A[M,K] * B[N,K]^T, bf16 in, fp32 acc ---------------
// EPI 0: store bf16
// EPI 1: +bias, ReLU, store bf16
// EPI 2: +res, store fp32
// EPI 3: +bias, +res, store fp32
// Block: 256 threads = 8 waves (2Mx4N); wave tile 64x32; block tile 128x128.
// K-slab 64 deep, double-buffered: A/B slabs 16KB each -> 64KB LDS total.

template <int EPI>
__global__ __launch_bounds__(256)
void gemm_nt(const __bf16* __restrict__ A, const __bf16* __restrict__ Bw,
             const float* __restrict__ bias, const float* __restrict__ res,
             void* __restrict__ outp, int M, int N, int K) {
    __shared__ __align__(16) __bf16 Ash[2][128 * 64];
    __shared__ __align__(16) __bf16 Bsh[2][128 * 64];

    const int tid  = threadIdx.x;
    const int lane = tid & 31;
    const int wid  = tid >> 5;
    const int mblk = blockIdx.y * 128;
    const int nblk = blockIdx.x * 128;
    const int m0 = (wid >> 2) * 64;   // wave row offset within block tile
    const int n0 = (wid & 3) * 32;    // wave col offset within block tile

    const unsigned aBase = lds_offset_of(&Ash[0][0]);
    const unsigned bBase = lds_offset_of(&Bsh[0][0]);

    // slab = 128 rows x 64 halves = 1024 x 16B chunks; 4 chunks per thread,
    // consecutive chunks per thread -> 64B contiguous global reads.
    auto issue = [&](int buf, int k0) {
#pragma unroll
        for (int i = 0; i < 4; i++) {
            const int c  = tid * 4 + i;
            const int r  = c >> 3;          // row 0..127
            const int kc = (c & 7) * 8;     // k offset in halves
            async_copy_b128(aBase + buf * 16384 + c * 16,
                            A + (size_t)(mblk + r) * K + k0 + kc);
            async_copy_b128(bBase + buf * 16384 + c * 16,
                            Bw + (size_t)(nblk + r) * K + k0 + kc);
        }
    };

    v8f acc[4][2] = {{v8f{}, v8f{}}, {v8f{}, v8f{}},
                     {v8f{}, v8f{}}, {v8f{}, v8f{}}};

    issue(0, 0);
    int buf = 0;
    for (int k0 = 0; k0 < K; k0 += 64, buf ^= 1) {
        const bool pre = (k0 + 64 < K);
        if (pre) {
            issue(buf ^ 1, k0 + 64);        // prefetch next slab
            wait_asynccnt<8>();             // current slab landed (in-order)
        } else {
            wait_asynccnt<0>();
        }
        __syncthreads();                    // all waves' copies visible

        const __bf16* As = &Ash[buf][0];
        const __bf16* Bs = &Bsh[buf][0];
#pragma unroll
        for (int ks = 0; ks < 2; ks++) {
            v16bf b0 = lds_tile64(Bs, n0,      ks, lane);
            v16bf b1 = lds_tile64(Bs, n0 + 16, ks, lane);
#pragma unroll
            for (int mt = 0; mt < 4; mt++) {
                v16bf a = lds_tile64(As, m0 + mt * 16, ks, lane);
                acc[mt][0] = wmma_bf16(a, b0, acc[mt][0]);
                acc[mt][1] = wmma_bf16(a, b1, acc[mt][1]);
            }
        }
        __syncthreads();                    // slab consumed, safe to overwrite
    }

    const int half8 = (lane >> 4) << 3;
#pragma unroll
    for (int mt = 0; mt < 4; mt++) {
#pragma unroll
        for (int nt = 0; nt < 2; nt++) {
#pragma unroll
            for (int j = 0; j < 8; j++) {
                const int row = mblk + m0 + mt * 16 + j + half8;
                const int col = nblk + n0 + nt * 16 + (lane & 15);
                const size_t idx = (size_t)row * N + col;
                float v = acc[mt][nt][j];
                if constexpr (EPI == 0) {
                    ((__bf16*)outp)[idx] = f2bf(v);
                } else if constexpr (EPI == 1) {
                    v += bias[col];
                    v = fmaxf(v, 0.0f);
                    ((__bf16*)outp)[idx] = f2bf(v);
                } else if constexpr (EPI == 2) {
                    ((float*)outp)[idx] = v + res[idx];
                } else {
                    ((float*)outp)[idx] = v + bias[col] + res[idx];
                }
            }
        }
    }
}

// ---- V transpose: v[B,S,H,DH] (bf16) -> vt[B,H,DH,S] (bf16) ---------------

__global__ void transpose_v(const unsigned short* __restrict__ v,
                            unsigned short* __restrict__ vt) {
    const int total = Bc * Hc * DHc * Sc;
    for (int i = blockIdx.x * blockDim.x + threadIdx.x; i < total;
         i += gridDim.x * blockDim.x) {
        const int s  = i & (Sc - 1);
        int t = i >> 11;                 // Sc = 2^11
        const int dh = t & (DHc - 1);
        t >>= 6;                          // DHc = 2^6
        const int h = t & (Hc - 1);
        const int b = t >> 4;             // Hc = 2^4
        vt[i] = v[(size_t)(b * Sc + s) * Dc + h * DHc + dh];
    }
}

// ---- Flash-style causal attention -----------------------------------------
// One wave per 16-row q-tile per (b,h). Online softmax, P relayout via a
// per-wave 1 KB LDS round trip (DS ops are in-order within a wave).

__global__ __launch_bounds__(256)
void attn_kernel(const __bf16* __restrict__ q, const __bf16* __restrict__ k,
                 const __bf16* __restrict__ vt, __bf16* __restrict__ out) {
    __shared__ unsigned short pshm[8][16 * 32];
    const int lane  = threadIdx.x & 31;
    const int wid   = threadIdx.x >> 5;
    const int qtile = blockIdx.x * 8 + wid;     // 0..127
    const int q0    = qtile * 16;
    const int b     = blockIdx.y / Hc;
    const int h     = blockIdx.y % Hc;
    const int half8 = (lane >> 4) << 3;

    const __bf16* qbase = q  + (size_t)b * Sc * Dc + h * DHc;
    const __bf16* kbase = k  + (size_t)b * Sc * Dc + h * DHc;
    const __bf16* vbase = vt + (size_t)(b * Hc + h) * DHc * Sc;

    // Q operand tiles (16 rows x 64 dh -> two 32-wide k-steps), loaded once
    const v16bf aq0 = load_nt(qbase, q0, Dc, 0,  lane);
    const v16bf aq1 = load_nt(qbase, q0, Dc, 32, lane);

    v8f oacc[4] = {v8f{}, v8f{}, v8f{}, v8f{}};
    float mrow[8], lrow[8];
#pragma unroll
    for (int j = 0; j < 8; j++) { mrow[j] = -1e30f; lrow[j] = 0.0f; }

    volatile unsigned short* pw = pshm[wid];

    for (int c0 = 0; c0 < q0 + 16; c0 += 32) {
        // scores: S[16 x 32] = Q(16x64) * K(c0..c0+31, 64)^T, two 16x16 tiles
        v8f s0 = {}, s1 = {};
        {
            v16bf bk = load_nt(kbase, c0, Dc, 0, lane);
            s0 = wmma_bf16(aq0, bk, s0);
            bk = load_nt(kbase, c0, Dc, 32, lane);
            s0 = wmma_bf16(aq1, bk, s0);
            bk = load_nt(kbase, c0 + 16, Dc, 0, lane);
            s1 = wmma_bf16(aq0, bk, s1);
            bk = load_nt(kbase, c0 + 16, Dc, 32, lane);
            s1 = wmma_bf16(aq1, bk, s1);
        }
        const bool need_mask = (c0 + 31 > q0);
#pragma unroll
        for (int j = 0; j < 8; j++) {
            const int row  = q0 + j + half8;
            const int col0 = c0 + (lane & 15);
            const int col1 = col0 + 16;
            float v0 = s0[j] * 0.125f;          // 1/sqrt(DH)
            float v1 = s1[j] * 0.125f;
            if (need_mask) {
                if (col0 > row) v0 = -1e30f;
                if (col1 > row) v1 = -1e30f;
            }
            const float tmax = rowmax16(fmaxf(v0, v1));
            const float mnew = fmaxf(mrow[j], tmax);
            const float sc   = __expf(mrow[j] - mnew);
            mrow[j] = mnew;
            const float p0 = __expf(v0 - mnew);
            const float p1 = __expf(v1 - mnew);
            lrow[j] = lrow[j] * sc + rowsum16(p0 + p1);
#pragma unroll
            for (int t2 = 0; t2 < 4; t2++) oacc[t2][j] *= sc;
            // stash P (C-layout) into LDS for A-operand relayout
            pw[(j + half8) * 32 + (lane & 15)]      = bf16bits(p0);
            pw[(j + half8) * 32 + 16 + (lane & 15)] = bf16bits(p1);
        }
        // reload P in WMMA A-operand layout (16 x 32 keys)
        union { unsigned short u[16]; v16bf v; } pa;
        const int r = lane & 15;
#pragma unroll
        for (int i = 0; i < 8; i++) {
            pa.u[i]     = pw[r * 32 + half8 + i];
            pa.u[8 + i] = pw[r * 32 + half8 + 16 + i];
        }
        // O += P(16x32) * V(32 keys x 64 dh): V^T columns are contiguous
#pragma unroll
        for (int nt = 0; nt < 4; nt++) {
            v16bf bv = load_nt(vbase, nt * 16, Sc, c0, lane);
            oacc[nt] = wmma_bf16(pa.v, bv, oacc[nt]);
        }
    }

#pragma unroll
    for (int j = 0; j < 8; j++) {
        const float inv = 1.0f / lrow[j];
        const int row = q0 + j + half8;
#pragma unroll
        for (int nt = 0; nt < 4; nt++) {
            const int dh = nt * 16 + (lane & 15);
            out[(size_t)(b * Sc + row) * Dc + h * DHc + dh] =
                f2bf(oacc[nt][j] * inv);
        }
    }
}

// ---------------------------------------------------------------------------

extern "C" void kernel_launch(void* const* d_in, const int* in_sizes, int n_in,
                              void* d_out, int out_size, void* d_ws, size_t ws_size,
                              hipStream_t stream) {
    (void)in_sizes; (void)n_in; (void)out_size; (void)ws_size;

    const float* x     = (const float*)d_in[0];
    // d_in[1] = mask (unused: causal mask applied analytically)
    const float* ln1_g = (const float*)d_in[2];
    const float* ln1_b = (const float*)d_in[3];
    const float* Wq    = (const float*)d_in[4];
    const float* Wk    = (const float*)d_in[5];
    const float* Wv    = (const float*)d_in[6];
    const float* Wo    = (const float*)d_in[7];
    const float* ln2_g = (const float*)d_in[8];
    const float* ln2_b = (const float*)d_in[9];
    const float* W1    = (const float*)d_in[10];
    const float* b1    = (const float*)d_in[11];
    const float* W2    = (const float*)d_in[12];
    const float* b2    = (const float*)d_in[13];

    // workspace carve-out (256B aligned slices)
    char* w = (char*)d_ws;
    size_t off = 0;
    auto alloc = [&](size_t bytes) -> void* {
        off = (off + 255) & ~(size_t)255;
        void* p = w + off;
        off += bytes;
        return p;
    };
    const size_t actE = (size_t)Mrows * Dc;          // 4M elements
    __bf16* xn   = (__bf16*)alloc(actE * 2);
    __bf16* qb   = (__bf16*)alloc(actE * 2);
    __bf16* kb   = (__bf16*)alloc(actE * 2);
    __bf16* vb   = (__bf16*)alloc(actE * 2);
    __bf16* vtb  = (__bf16*)alloc(actE * 2);
    __bf16* ao   = (__bf16*)alloc(actE * 2);         // attention output
    float*  x2   = (float*)alloc(actE * 4);          // post-attention residual
    __bf16* hn   = (__bf16*)alloc(actE * 2);         // LN2 output
    __bf16* hmid = (__bf16*)alloc((size_t)Mrows * DFFc * 2);
    __bf16* wqb  = (__bf16*)alloc((size_t)Dc * Dc * 2);
    __bf16* wkb  = (__bf16*)alloc((size_t)Dc * Dc * 2);
    __bf16* wvb  = (__bf16*)alloc((size_t)Dc * Dc * 2);
    __bf16* wob  = (__bf16*)alloc((size_t)Dc * Dc * 2);
    __bf16* w1b  = (__bf16*)alloc((size_t)DFFc * Dc * 2);
    __bf16* w2b  = (__bf16*)alloc((size_t)Dc * DFFc * 2);

    // 1) weight converts (fp32 -> bf16)
    cvt_kernel<<<512, 256, 0, stream>>>(Wq, wqb, Dc * Dc);
    cvt_kernel<<<512, 256, 0, stream>>>(Wk, wkb, Dc * Dc);
    cvt_kernel<<<512, 256, 0, stream>>>(Wv, wvb, Dc * Dc);
    cvt_kernel<<<512, 256, 0, stream>>>(Wo, wob, Dc * Dc);
    cvt_kernel<<<2048, 256, 0, stream>>>(W1, w1b, DFFc * Dc);
    cvt_kernel<<<2048, 256, 0, stream>>>(W2, w2b, Dc * DFFc);

    // 2) LN1: x -> xn (bf16)
    ln_kernel<<<Mrows, 256, 0, stream>>>(x, ln1_g, ln1_b, xn);

    // 3) Q/K/V projections: [4096,1024] = xn * W^T
    {
        dim3 g(Dc / 128, Mrows / 128);
        gemm_nt<0><<<g, 256, 0, stream>>>(xn, wqb, nullptr, nullptr, qb,
                                          Mrows, Dc, Dc);
        gemm_nt<0><<<g, 256, 0, stream>>>(xn, wkb, nullptr, nullptr, kb,
                                          Mrows, Dc, Dc);
        gemm_nt<0><<<g, 256, 0, stream>>>(xn, wvb, nullptr, nullptr, vb,
                                          Mrows, Dc, Dc);
    }

    // 4) V transpose for contiguous PV B-operand columns
    transpose_v<<<8192, 256, 0, stream>>>((const unsigned short*)vb,
                                          (unsigned short*)vtb);

    // 5) causal attention -> ao (bf16)
    {
        dim3 g(Sc / 16 / 8, Bc * Hc);  // (16, 32), 8 q-tile waves per block
        attn_kernel<<<g, 256, 0, stream>>>(qb, kb, vtb, ao);
    }

    // 6) O-projection + residual: x2 = x + ao * Wo^T   (fp32)
    {
        dim3 g(Dc / 128, Mrows / 128);
        gemm_nt<2><<<g, 256, 0, stream>>>(ao, wob, nullptr, x, x2,
                                          Mrows, Dc, Dc);
    }

    // 7) LN2: x2 -> hn (bf16)
    ln_kernel<<<Mrows, 256, 0, stream>>>(x2, ln2_g, ln2_b, hn);

    // 8) FFN1: hmid = relu(hn * W1^T + b1) (bf16)
    {
        dim3 g(DFFc / 128, Mrows / 128);
        gemm_nt<1><<<g, 256, 0, stream>>>(hn, w1b, b1, nullptr, hmid,
                                          Mrows, DFFc, Dc);
    }

    // 9) FFN2 + bias + residual: out = x2 + hmid * W2^T + b2 (fp32)
    {
        dim3 g(Dc / 128, Mrows / 128);
        gemm_nt<3><<<g, 256, 0, stream>>>(hmid, w2b, b2, x2, (float*)d_out,
                                          Mrows, Dc, DFFc);
    }
}
